// PointerNet_1520418422846
// MI455X (gfx1250) — compile-verified, matrix operand order checked by reading, not run on hardware
//
#include <hip/hip_runtime.h>
#include <hip/hip_bf16.h>
#include <cstdint>

#define B_   1024
#define S_   128
#define E_   256
#define H_   256
#define XH_  512      // concat [x | h]
#define G4H_ 1024     // 4*H gates
#define NEGV (-1e9f)

typedef __bf16 bf16;
typedef __attribute__((ext_vector_type(16))) __bf16 v16bf;
typedef __attribute__((ext_vector_type(8)))  __bf16 v8bf;
typedef __attribute__((ext_vector_type(8)))  float   v8f;
typedef __attribute__((ext_vector_type(4)))  unsigned v4u;
typedef __attribute__((ext_vector_type(8)))  int      v8i;
typedef __attribute__((ext_vector_type(4)))  int      v4i;

// ----- CDNA5 data-mover path selection --------------------------------------
#if __has_builtin(__builtin_amdgcn_tensor_load_to_lds)
#define USE_TDM 1
#elif __has_builtin(__builtin_amdgcn_global_load_async_to_lds_b128)
#define USE_ASYNC 1
#endif

#define BM 128
#define BN 128
#define BK 32
#define LDS_STRIDE 40            // padded row stride (elements) -> 80B
#define TILE (BM * LDS_STRIDE)   // elements per staged tile
#define TILE_BYTES (TILE * 2)

#if defined(USE_TDM)
// Issue one TDM load of a [128 rows x 32 cols] bf16 tile into LDS.
// TDM padding: pad_interval=3 (16 DWORDs = 64B = one 32-elem row),
// pad_amount=3 (4 DWORDs = 16B) -> LDS row stride 80B == LDS_STRIDE elems.
__device__ __forceinline__ void tdm_tile_load(const bf16* gtile, unsigned lds_off,
                                              int lda_elems, int tensor_rows)
{
    unsigned long long ga = (unsigned long long)(uintptr_t)gtile;
    v4u g0;
    g0[0] = 1u;                                       // count=1, user mode
    g0[1] = lds_off;                                  // lds_addr (bytes)
    g0[2] = (unsigned)ga;                             // global_addr[31:0]
    g0[3] = (unsigned)((ga >> 32) & 0x1FFFFFFu) | (2u << 30);  // addr[56:32] | type=2
    v8i g1;
    g1[0] = (1 << 16) | (1 << 20) | (3 << 22) | (3 << 25);  // 2B elems, pad 16DW/+4DW
    g1[1] = (int)(((unsigned)lda_elems & 0xffffu) << 16);   // tensor_dim0[15:0]
    g1[2] = (int)((((unsigned)lda_elems >> 16) & 0xffffu)
                  | (((unsigned)tensor_rows & 0xffffu) << 16)); // dim0 hi | dim1 lo
    g1[3] = (int)((((unsigned)tensor_rows >> 16) & 0xffffu)
                  | ((unsigned)BK << 16));                       // dim1 hi | tile_dim0
    g1[4] = BM;                                       // tile_dim1 (tile_dim2=0)
    g1[5] = lda_elems;                                // tensor_dim0_stride[31:0]
    g1[6] = 0;
    g1[7] = 0;
    v4i gz = {0, 0, 0, 0};
#if __clang_major__ >= 23
    v8i gz8 = {0, 0, 0, 0, 0, 0, 0, 0};
    __builtin_amdgcn_tensor_load_to_lds(g0, g1, gz, gz, gz8, 0);
#else
    __builtin_amdgcn_tensor_load_to_lds(g0, g1, gz, gz, 0);
#endif
}
#endif

// ---------------------------------------------------------------------------
// WMMA GEMM: C[M,N] = A[M,K] @ W[N,K]^T + bias,  A/W bf16, accum f32.
// Block tile 128x128, BK=32, 256 threads (8 waves, 2x4 wave grid,
// each wave does 4 M-tiles x 2 N-tiles of v_wmma_f32_16x16x32_bf16).
// Tile staging: double-buffered TDM tensor_load_to_lds (wave 0) overlapping
// the WMMA compute -> ASYNC global->LDS -> manual fallback.
// ---------------------------------------------------------------------------
__global__ __launch_bounds__(256, 2)
void k_gemm_bf16(const bf16* __restrict__ A, int lda,
                 const bf16* __restrict__ W, int ldw,
                 const float* __restrict__ bias,
                 float* __restrict__ Cf, bf16* __restrict__ Cb, int ldc,
                 int M, int N, int K)
{
#if defined(USE_TDM)
    __shared__ __attribute__((aligned(16))) bf16 As[2 * TILE];
    __shared__ __attribute__((aligned(16))) bf16 Bs[2 * TILE];
#else
    __shared__ __attribute__((aligned(16))) bf16 As[TILE];
    __shared__ __attribute__((aligned(16))) bf16 Bs[TILE];
#endif

    const int tid   = threadIdx.x;
    const int wave  = tid >> 5;
    const int lane  = tid & 31;
    const int waveM = wave >> 2;      // 0..1 (64 rows each)
    const int waveN = wave & 3;       // 0..3 (32 cols each)
    const int blockM = blockIdx.y * BM;
    const int blockN = blockIdx.x * BN;

    const v8f vzero = {0.f,0.f,0.f,0.f,0.f,0.f,0.f,0.f};
    v8f acc[4][2];
#pragma unroll
    for (int i = 0; i < 4; ++i)
#pragma unroll
        for (int j = 0; j < 2; ++j) acc[i][j] = vzero;

    const int lrow = tid >> 2;        // 0..63
    const int lseg = tid & 3;         // 0..3 (8 bf16 = 16B each)
    const int half = lane >> 4;       // hi/lo K-half per ISA A/B layout
    const int l16  = lane & 15;

    const int nK = K / BK;

#if defined(USE_TDM)
    // prologue: stage tile 0 into buffer 0
    if (tid < 32) {
        tdm_tile_load(A + (size_t)blockM * lda, (unsigned)(uintptr_t)As, lda, M);
        tdm_tile_load(W + (size_t)blockN * ldw, (unsigned)(uintptr_t)Bs, ldw, N);
    }
#endif

    for (int kt = 0; kt < nK; ++kt) {
        const int k0 = kt * BK;
#if defined(USE_TDM)
        const int cur = kt & 1;
        if (tid < 32) {   // wave 0 drives the Tensor Data Mover
            if (kt + 1 < nK) {
                // issue next tile into the other buffer, then wait until only
                // that just-issued pair is outstanding -> current tile resident
                tdm_tile_load(A + (size_t)blockM * lda + k0 + BK,
                              (unsigned)(uintptr_t)As + (unsigned)((cur ^ 1) * TILE_BYTES),
                              lda, M);
                tdm_tile_load(W + (size_t)blockN * ldw + k0 + BK,
                              (unsigned)(uintptr_t)Bs + (unsigned)((cur ^ 1) * TILE_BYTES),
                              ldw, N);
                __builtin_amdgcn_s_wait_tensorcnt(2);
            } else {
                __builtin_amdgcn_s_wait_tensorcnt(0);
            }
        }
        __syncthreads();
        const bf16* Ac = As + cur * TILE;
        const bf16* Bc = Bs + cur * TILE;
        if (k0 + 2 * BK < K) {  // L2 warm-up for the tile after next
            __builtin_prefetch(A + (size_t)(blockM + lrow) * lda + k0 + 2 * BK, 0, 1);
            __builtin_prefetch(W + (size_t)(blockN + lrow) * ldw + k0 + 2 * BK, 0, 1);
        }
#elif defined(USE_ASYNC)
#pragma unroll
        for (int rr = 0; rr < BM; rr += 64) {
            const int r = lrow + rr;
            __builtin_amdgcn_global_load_async_to_lds_b128(
                (__attribute__((address_space(1))) const void*)(uintptr_t)
                    (A + (size_t)(blockM + r) * lda + k0 + lseg * 8),
                (__attribute__((address_space(3))) void*)(unsigned)(uintptr_t)
                    (As + r * LDS_STRIDE + lseg * 8),
                0, 0);
            __builtin_amdgcn_global_load_async_to_lds_b128(
                (__attribute__((address_space(1))) const void*)(uintptr_t)
                    (W + (size_t)(blockN + r) * ldw + k0 + lseg * 8),
                (__attribute__((address_space(3))) void*)(unsigned)(uintptr_t)
                    (Bs + r * LDS_STRIDE + lseg * 8),
                0, 0);
        }
        asm volatile("s_wait_asynccnt 0" ::: "memory");
        __syncthreads();
        const bf16* Ac = As;
        const bf16* Bc = Bs;
        if (k0 + BK < K) {
            __builtin_prefetch(A + (size_t)(blockM + lrow) * lda + k0 + BK, 0, 1);
            __builtin_prefetch(W + (size_t)(blockN + lrow) * ldw + k0 + BK, 0, 1);
        }
#else
#pragma unroll
        for (int rr = 0; rr < BM; rr += 64) {
            const int r = lrow + rr;
            *(v8bf*)(As + r * LDS_STRIDE + lseg * 8) =
                *(const v8bf*)(A + (size_t)(blockM + r) * lda + k0 + lseg * 8);
            *(v8bf*)(Bs + r * LDS_STRIDE + lseg * 8) =
                *(const v8bf*)(W + (size_t)(blockN + r) * ldw + k0 + lseg * 8);
        }
        __syncthreads();
        const bf16* Ac = As;
        const bf16* Bc = Bs;
        if (k0 + BK < K) {
            __builtin_prefetch(A + (size_t)(blockM + lrow) * lda + k0 + BK, 0, 1);
            __builtin_prefetch(W + (size_t)(blockN + lrow) * ldw + k0 + BK, 0, 1);
        }
#endif

        // ---- build fragments (16 bf16/lane = two ds_load_b128 per tile) ----
        v16bf afrag[4];
#pragma unroll
        for (int mt = 0; mt < 4; ++mt) {
            const int m = waveM * 64 + mt * 16 + l16;
            v8bf lo = *(const v8bf*)(Ac + m * LDS_STRIDE + half * 8);
            v8bf hi = *(const v8bf*)(Ac + m * LDS_STRIDE + 16 + half * 8);
            v16bf f;
#pragma unroll
            for (int e = 0; e < 8; ++e) { f[e] = lo[e]; f[8 + e] = hi[e]; }
            afrag[mt] = f;
        }
        v16bf bfrag[2];
#pragma unroll
        for (int nt = 0; nt < 2; ++nt) {
            const int n = waveN * 32 + nt * 16 + l16;
            v8bf lo = *(const v8bf*)(Bc + n * LDS_STRIDE + half * 8);
            v8bf hi = *(const v8bf*)(Bc + n * LDS_STRIDE + 16 + half * 8);
            v16bf f;
#pragma unroll
            for (int e = 0; e < 8; ++e) { f[e] = lo[e]; f[8 + e] = hi[e]; }
            bfrag[nt] = f;
        }

        // ---- 8 WMMAs per wave per K-step ----
#pragma unroll
        for (int mt = 0; mt < 4; ++mt)
#pragma unroll
            for (int nt = 0; nt < 2; ++nt)
                acc[mt][nt] = __builtin_amdgcn_wmma_f32_16x16x32_bf16(
                    false, afrag[mt], false, bfrag[nt],
                    (short)0, acc[mt][nt], false, false);
        __syncthreads();   // protects the buffer overwritten next iteration
    }

    // ---- epilogue: f32 C layout (VGPR v, lane L): M=v+(L>>4)*8, N=L&15 ----
    const int mhalf = (lane >> 4) * 8;
#pragma unroll
    for (int mt = 0; mt < 4; ++mt) {
#pragma unroll
        for (int nt = 0; nt < 2; ++nt) {
            const int col = blockN + waveN * 32 + nt * 16 + l16;
            const float bv = bias ? bias[col] : 0.f;
#pragma unroll
            for (int v = 0; v < 8; ++v) {
                const int row = blockM + waveM * 64 + mt * 16 + mhalf + v;
                const float val = acc[mt][nt][v] + bv;
                if (Cf) Cf[(size_t)row * ldc + col] = val;
                if (Cb) Cb[(size_t)row * ldc + col] = (bf16)val;
            }
        }
    }
}

// ---------------------------------------------------------------------------
// Elementwise / small kernels
// ---------------------------------------------------------------------------
__global__ void k_embed(const float* __restrict__ inp,   // [B,2,S]
                        const float* __restrict__ Wemb,  // [2,E]
                        float* __restrict__ outf,        // [B,S,E] (d_out)
                        bf16* __restrict__ outb)         // bf16 copy (ws)
{
    const int idx = blockIdx.x * blockDim.x + threadIdx.x;   // B*S*E
    const int e = idx & 255;
    const int s = (idx >> 8) & 127;
    const int b = idx >> 15;
    const float x0 = inp[(size_t)b * 256 + s];
    const float x1 = inp[(size_t)b * 256 + 128 + s];
    const float v = x0 * Wemb[e] + x1 * Wemb[256 + e];
    outf[idx] = v;
    outb[idx] = (bf16)v;
}

__global__ void k_cvt(const float* __restrict__ src, bf16* __restrict__ dst,
                      int cols, int dstStride, int dstOff, int n)
{
    const int idx = blockIdx.x * blockDim.x + threadIdx.x;
    if (idx >= n) return;
    const int r = idx / cols, c = idx % cols;
    dst[(size_t)r * dstStride + dstOff + c] = (bf16)src[idx];
}

__global__ void k_bias_cat(const float* __restrict__ a, const float* __restrict__ b,
                           float* __restrict__ o, int n)
{
    const int idx = blockIdx.x * blockDim.x + threadIdx.x;
    if (idx < n) o[idx] = a[idx] + b[idx];
}

__global__ void k_load_x(const bf16* __restrict__ emb, int t, bf16* __restrict__ xh)
{
    const int idx = blockIdx.x * blockDim.x + threadIdx.x;   // B*E
    const int b = idx >> 8, e = idx & 255;
    xh[(size_t)b * XH_ + e] = emb[((size_t)b * S_ + t) * E_ + e];
}

__global__ void k_set_decstart(const float* __restrict__ ds, bf16* __restrict__ xh)
{
    const int idx = blockIdx.x * blockDim.x + threadIdx.x;   // B*E
    const int b = idx >> 8, e = idx & 255;
    xh[(size_t)b * XH_ + e] = (bf16)ds[e];
}

__global__ void k_lstm_update(const float* __restrict__ gates,  // [B,4H] i,f,g,o
                              float* __restrict__ c,            // [B,H]
                              bf16* __restrict__ xh,            // h -> xh[:,256:]
                              bf16* __restrict__ hcopy, long hstride)
{
    const int idx = blockIdx.x * blockDim.x + threadIdx.x;     // B*H
    const int b = idx >> 8, h = idx & 255;
    const float* g = gates + (size_t)b * G4H_;
    const float gi = g[h], gf = g[256 + h], gg = g[512 + h], go = g[768 + h];
    const float si = 1.f / (1.f + __expf(-gi));
    const float sf = 1.f / (1.f + __expf(-gf));
    const float so = 1.f / (1.f + __expf(-go));
    const float cn = sf * c[idx] + si * tanhf(gg);
    c[idx] = cn;
    const float hn = so * tanhf(cn);
    xh[(size_t)b * XH_ + 256 + h] = (bf16)hn;
    if (hcopy) hcopy[(size_t)b * hstride + h] = (bf16)hn;
}

// one wave per (b,s): lg = C * tanh(V . tanh(qq[b] + ref[b,s])), with masking
__global__ void k_attn_logits(const float* __restrict__ qq,     // [B,H]
                              const bf16* __restrict__ ref,     // [B,S,H]
                              const float* __restrict__ V,      // [H]
                              unsigned char* __restrict__ mask, // [B,S]
                              const int* __restrict__ idxs,     // [B]
                              int t, float* __restrict__ logits)
{
    const int wid  = blockIdx.x * (blockDim.x >> 5) + (threadIdx.x >> 5);
    const int lane = threadIdx.x & 31;
    const int b = wid >> 7, s = wid & 127;
    const bf16* r = ref + ((size_t)b * S_ + s) * H_;
    const float* q = qq + (size_t)b * H_;
    float acc = 0.f;
#pragma unroll
    for (int e = 0; e < 8; ++e) {
        const int h = lane * 8 + e;
        acc += V[h] * tanhf(q[h] + (float)r[h]);
    }
#pragma unroll
    for (int off = 16; off > 0; off >>= 1) acc += __shfl_xor(acc, off, 32);
    if (lane == 0) {
        if (t > 0 && idxs[b] == s) mask[(size_t)b * S_ + s] = 1;
        logits[(size_t)b * S_ + s] =
            mask[(size_t)b * S_ + s] ? NEGV : 10.0f * tanhf(acc);
    }
}

// block per b: softmax over S, then context[h] = sum_s w[s]*ref[b,s,h]
__global__ void k_softmax_context(const float* __restrict__ logits,
                                  const bf16* __restrict__ ref,
                                  float* __restrict__ ctx, bf16* __restrict__ ctxb)
{
    __shared__ float w[S_];
    __shared__ float red[256];
    const int b = blockIdx.x, tid = threadIdx.x;
    const float lv = (tid < S_) ? logits[(size_t)b * S_ + tid] : -1e30f;
    red[tid] = lv; __syncthreads();
    for (int off = 128; off > 0; off >>= 1) {
        if (tid < off) red[tid] = fmaxf(red[tid], red[tid + off]);
        __syncthreads();
    }
    const float mx = red[0]; __syncthreads();
    const float ev = (tid < S_) ? __expf(lv - mx) : 0.f;
    red[tid] = ev; __syncthreads();
    for (int off = 128; off > 0; off >>= 1) {
        if (tid < off) red[tid] += red[tid + off];
        __syncthreads();
    }
    const float inv = 1.f / red[0];
    if (tid < S_) w[tid] = ev * inv;
    __syncthreads();
    const int h = tid;                       // H_ == blockDim.x == 256
    float cacc = 0.f;
    const bf16* base = ref + (size_t)b * S_ * H_ + h;
    for (int s = 0; s < S_; ++s) cacc += w[s] * (float)base[(size_t)s * H_];
    ctx[(size_t)b * H_ + h]  = cacc;
    ctxb[(size_t)b * H_ + h] = (bf16)cacc;
}

// block per b (128 thr): softmax -> probs out, hash-categorical sample,
// gather next decoder input from embedded (bf16) into xh x-half.
__global__ void k_softmax_sample(const float* __restrict__ logits, int t,
                                 float* __restrict__ probs_out,   // [B,S] slice
                                 int* __restrict__ idx_out,       // [B] slice
                                 int* __restrict__ idx_state,     // [B]
                                 const bf16* __restrict__ emb,    // [B,S,E]
                                 bf16* __restrict__ xh)
{
    __shared__ float p[S_];
    __shared__ float red[S_];
    __shared__ int chosen;
    const int b = blockIdx.x, tid = threadIdx.x;
    const float lv = logits[(size_t)b * S_ + tid];
    red[tid] = lv; __syncthreads();
    for (int off = 64; off > 0; off >>= 1) {
        if (tid < off) red[tid] = fmaxf(red[tid], red[tid + off]);
        __syncthreads();
    }
    const float mx = red[0]; __syncthreads();
    const float ev = __expf(lv - mx);
    red[tid] = ev; __syncthreads();
    for (int off = 64; off > 0; off >>= 1) {
        if (tid < off) red[tid] += red[tid + off];
        __syncthreads();
    }
    const float pr = ev / red[0];
    p[tid] = pr;
    probs_out[(size_t)b * S_ + tid] = pr;
    __syncthreads();
    if (tid == 0) {
        unsigned x = (unsigned)(t * B_ + b) * 2654435761u + 0x9e3779b9u;
        x ^= x << 13; x ^= x >> 17; x ^= x << 5;
        const float u = (float)(x >> 8) * (1.0f / 16777216.0f);
        float cs = 0.f; int sel = S_ - 1;
        for (int s = 0; s < S_; ++s) { cs += p[s]; if (u < cs) { sel = s; break; } }
        chosen = sel;
        idx_out[b]   = sel;
        idx_state[b] = sel;
    }
    __syncthreads();
    const int sel = chosen;
    const bf16* src = emb + ((size_t)b * S_ + sel) * E_;
    xh[(size_t)b * XH_ + tid]        = src[tid];
    xh[(size_t)b * XH_ + tid + 128]  = src[tid + 128];
}

// ---------------------------------------------------------------------------
extern "C" void kernel_launch(void* const* d_in, const int* in_sizes, int n_in,
                              void* d_out, int out_size, void* d_ws, size_t ws_size,
                              hipStream_t stream)
{
    const float* in_inputs  = (const float*)d_in[0];
    const float* in_embW    = (const float*)d_in[1];
    const float* enc_Wih    = (const float*)d_in[2];
    const float* enc_Whh    = (const float*)d_in[3];
    const float* enc_bih    = (const float*)d_in[4];
    const float* enc_bhh    = (const float*)d_in[5];
    const float* dec_Wih    = (const float*)d_in[6];
    const float* dec_Whh    = (const float*)d_in[7];
    const float* dec_bih    = (const float*)d_in[8];
    const float* dec_bhh    = (const float*)d_in[9];
    const float* dec_start  = (const float*)d_in[10];
    const float* glm_Wq     = (const float*)d_in[11];
    const float* glm_bq     = (const float*)d_in[12];
    const float* glm_Wref   = (const float*)d_in[13];
    const float* glm_bref   = (const float*)d_in[14];
    const float* glm_V      = (const float*)d_in[15];
    const float* ptr_Wq     = (const float*)d_in[16];
    const float* ptr_bq     = (const float*)d_in[17];
    const float* ptr_Wref   = (const float*)d_in[18];
    const float* ptr_bref   = (const float*)d_in[19];
    const float* ptr_V      = (const float*)d_in[20];

    float* outf = (float*)d_out;
    float* probs_base = outf;                                         // [S,B,S]
    int*   idx_base   = (int*)(outf + (size_t)S_ * B_ * S_);          // [S,B]
    float* emb_out    = outf + (size_t)S_ * B_ * S_ + (size_t)S_ * B_; // [B,S,E]

    // -------- workspace carve-up (256B aligned) --------
    char* ws = (char*)d_ws;
    size_t off = 0;
    auto alloc = [&](size_t bytes) -> char* {
        char* p = ws + off;
        off = (off + bytes + 255) & ~(size_t)255;
        return p;
    };
    bf16* emb_b      = (bf16*)alloc((size_t)B_ * S_ * E_ * 2);
    bf16* enc_out_b  = (bf16*)alloc((size_t)B_ * S_ * H_ * 2);
    bf16* glm_ref_b  = (bf16*)alloc((size_t)B_ * S_ * H_ * 2);
    bf16* ptr_ref_b  = (bf16*)alloc((size_t)B_ * S_ * H_ * 2);
    bf16* wcat_enc   = (bf16*)alloc((size_t)G4H_ * XH_ * 2);
    bf16* wcat_dec   = (bf16*)alloc((size_t)G4H_ * XH_ * 2);
    bf16* wq_glm     = (bf16*)alloc((size_t)H_ * H_ * 2);
    bf16* wref_glm   = (bf16*)alloc((size_t)H_ * H_ * 2);
    bf16* wq_ptr     = (bf16*)alloc((size_t)H_ * H_ * 2);
    bf16* wref_ptr   = (bf16*)alloc((size_t)H_ * H_ * 2);
    float* bias_enc  = (float*)alloc((size_t)G4H_ * 4);
    float* bias_dec  = (float*)alloc((size_t)G4H_ * 4);
    bf16* xh         = (bf16*)alloc((size_t)B_ * XH_ * 2);
    float* gates     = (float*)alloc((size_t)B_ * G4H_ * 4);
    float* cbuf      = (float*)alloc((size_t)B_ * H_ * 4);
    float* qq        = (float*)alloc((size_t)B_ * H_ * 4);
    float* ctx       = (float*)alloc((size_t)B_ * H_ * 4);
    bf16* ctxb       = (bf16*)alloc((size_t)B_ * H_ * 2);
    float* logits    = (float*)alloc((size_t)B_ * S_ * 4);
    unsigned char* mask = (unsigned char*)alloc((size_t)B_ * S_);
    int*  idx_state  = (int*)alloc((size_t)B_ * 4);

    // -------- weight conversions (f32 -> bf16, with concatenation) --------
    const int nW = G4H_ * H_;    // 262144
    k_cvt<<<nW / 256, 256, 0, stream>>>(enc_Wih, wcat_enc, E_, XH_, 0,   nW);
    k_cvt<<<nW / 256, 256, 0, stream>>>(enc_Whh, wcat_enc, H_, XH_, E_, nW);
    k_cvt<<<nW / 256, 256, 0, stream>>>(dec_Wih, wcat_dec, E_, XH_, 0,   nW);
    k_cvt<<<nW / 256, 256, 0, stream>>>(dec_Whh, wcat_dec, H_, XH_, E_, nW);
    const int nQ = H_ * H_;      // 65536
    k_cvt<<<nQ / 256, 256, 0, stream>>>(glm_Wq,   wq_glm,   H_, H_, 0, nQ);
    k_cvt<<<nQ / 256, 256, 0, stream>>>(glm_Wref, wref_glm, H_, H_, 0, nQ);
    k_cvt<<<nQ / 256, 256, 0, stream>>>(ptr_Wq,   wq_ptr,   H_, H_, 0, nQ);
    k_cvt<<<nQ / 256, 256, 0, stream>>>(ptr_Wref, wref_ptr, H_, H_, 0, nQ);
    k_bias_cat<<<G4H_ / 256, 256, 0, stream>>>(enc_bih, enc_bhh, bias_enc, G4H_);
    k_bias_cat<<<G4H_ / 256, 256, 0, stream>>>(dec_bih, dec_bhh, bias_dec, G4H_);

    // -------- embedding --------
    k_embed<<<(B_ * S_ * E_) / 256, 256, 0, stream>>>(in_inputs, in_embW, emb_out, emb_b);

    // -------- encoder init: h = c = 0 --------
    hipMemsetAsync(xh,   0, (size_t)B_ * XH_ * 2, stream);
    hipMemsetAsync(cbuf, 0, (size_t)B_ * H_ * 4, stream);

    // -------- encoder LSTM (128 steps) --------
    for (int t = 0; t < S_; ++t) {
        k_load_x<<<(B_ * E_) / 256, 256, 0, stream>>>(emb_b, t, xh);
        k_gemm_bf16<<<dim3(G4H_ / BN, B_ / BM), 256, 0, stream>>>(
            xh, XH_, wcat_enc, XH_, bias_enc, gates, nullptr, G4H_, B_, G4H_, XH_);
        k_lstm_update<<<(B_ * H_) / 256, 256, 0, stream>>>(
            gates, cbuf, xh, enc_out_b + (size_t)t * H_, (long)S_ * H_);
    }

    // -------- step-invariant ref projections (bf16 out, bias fused) --------
    k_gemm_bf16<<<dim3(H_ / BN, (B_ * S_) / BM), 256, 0, stream>>>(
        enc_out_b, H_, wref_glm, H_, glm_bref, nullptr, glm_ref_b, H_, B_ * S_, H_, H_);
    k_gemm_bf16<<<dim3(H_ / BN, (B_ * S_) / BM), 256, 0, stream>>>(
        enc_out_b, H_, wref_ptr, H_, ptr_bref, nullptr, ptr_ref_b, H_, B_ * S_, H_, H_);

    // -------- decoder init (h,c carried from encoder; dec_in = dec_start) ---
    k_set_decstart<<<(B_ * E_) / 256, 256, 0, stream>>>(dec_start, xh);
    hipMemsetAsync(mask,      0, (size_t)B_ * S_, stream);
    hipMemsetAsync(idx_state, 0, (size_t)B_ * 4, stream);

    // -------- decoder loop (128 steps) --------
    for (int t = 0; t < S_; ++t) {
        // LSTM cell
        k_gemm_bf16<<<dim3(G4H_ / BN, B_ / BM), 256, 0, stream>>>(
            xh, XH_, wcat_dec, XH_, bias_dec, gates, nullptr, G4H_, B_, G4H_, XH_);
        k_lstm_update<<<(B_ * H_) / 256, 256, 0, stream>>>(
            gates, cbuf, xh, (bf16*)nullptr, 0L);

        // glimpse: qq = h @ Wq^T + bq  (A = h half of xh, lda = 512)
        k_gemm_bf16<<<dim3(H_ / BN, B_ / BM), 256, 0, stream>>>(
            xh + 256, XH_, wq_glm, H_, glm_bq, qq, nullptr, H_, B_, H_, H_);
        k_attn_logits<<<(B_ * S_) / 8, 256, 0, stream>>>(
            qq, glm_ref_b, glm_V, mask, idx_state, t, logits);
        k_softmax_context<<<B_, 256, 0, stream>>>(logits, glm_ref_b, ctx, ctxb);

        // pointer: qq = ctx @ Wq^T + bq
        k_gemm_bf16<<<dim3(H_ / BN, B_ / BM), 256, 0, stream>>>(
            ctxb, H_, wq_ptr, H_, ptr_bq, qq, nullptr, H_, B_, H_, H_);
        k_attn_logits<<<(B_ * S_) / 8, 256, 0, stream>>>(
            qq, ptr_ref_b, ptr_V, mask, idx_state, t, logits);

        // softmax -> probs out, sample, set next dec_in
        k_softmax_sample<<<B_, 128, 0, stream>>>(
            logits, t,
            probs_base + (size_t)t * B_ * S_,
            idx_base + (size_t)t * B_,
            idx_state, emb_b, xh);
    }
}
